// Kernel_2594160247416
// MI455X (gfx1250) — matrix operand order repaired and checked by reading, not compiled, not run on hardware
//
#include <hip/hip_runtime.h>
#include <math.h>
#include <stdint.h>

// ---------------------------------------------------------------------------
// CDNA5 / gfx1250 WMMA types
// ---------------------------------------------------------------------------
typedef __attribute__((ext_vector_type(16))) __bf16 v16bf;
typedef __attribute__((ext_vector_type(8)))  float  v8f;

#define F   256   // feature dim / kernel_size (L is FxF)
#define BM  128   // i-tile (rows of output, from support_y)
#define BN  64    // j-tile (cols of output, from support_x)
#define KT  32    // K step per WMMA
#define PAD 8     // LDS row padding (bf16 elems): stride 40*2B = 80B -> conflict-free frag reads
#define LROW (KT + PAD)

__device__ __forceinline__ unsigned short f32_to_bf16_bits(float f) {
  unsigned u = __float_as_uint(f);
  unsigned r = u + 0x7FFFu + ((u >> 16) & 1u);   // round-to-nearest-even
  return (unsigned short)(r >> 16);
}
__device__ __forceinline__ float bf16_bits_to_f32(unsigned short h) {
  return __uint_as_float(((unsigned)h) << 16);
}

union Frag { v16bf v; uint4 q[2]; };

// CDNA5 async global->LDS copy (GV mode, per-lane 16B), tracked by ASYNCcnt.
// lds_byte_off = low 32 bits of the generic address of the LDS destination
// (aperture mapping: flat addr[31:0] == wave-relative LDS offset).
__device__ __forceinline__ void async_copy_b128(unsigned lds_byte_off, const void* g) {
  asm volatile("global_load_async_to_lds_b128 %0, %1, off"
               :: "v"(lds_byte_off), "v"((unsigned long long)(uintptr_t)g)
               : "memory");
}

// ---------------------------------------------------------------------------
// Kernel 1: XL = X@L (and YL = Y@L), f32 accumulate, split to bf16 hi/lo,
// plus row norms ||XL_r||^2 (wave-shuffle reduce + f32 atomics).
// blockDim = 256 = F threads; each block: 16 rows, all 256 columns.
// ---------------------------------------------------------------------------
__global__ __launch_bounds__(256) void prep_gemm_split(
    const float* __restrict__ X, const float* __restrict__ Y,
    const float* __restrict__ L,
    unsigned short* __restrict__ XLh, unsigned short* __restrict__ XLl,
    unsigned short* __restrict__ YLh, unsigned short* __restrict__ YLl,
    float* __restrict__ xnorm, float* __restrict__ ynorm,
    int n1, int n2)
{
  const int isY     = blockIdx.y;
  const int rows    = isY ? n2 : n1;
  const int rowBase = blockIdx.x * 16;
  if (rowBase >= rows) return;
  const float* src = isY ? Y : X;
  unsigned short* dH = isY ? YLh : XLh;
  unsigned short* dL = isY ? YLl : XLl;
  float* nrm = isY ? ynorm : xnorm;

  __shared__ float Xs[16][F + 1];
  const int t = threadIdx.x;
  for (int n = t; n < 16 * F; n += 256) {
    int r = n >> 8, c = n & (F - 1);
    Xs[r][c] = src[(size_t)(rowBase + r) * F + c];
  }
  __syncthreads();

  float acc[16];
#pragma unroll
  for (int r = 0; r < 16; ++r) acc[r] = 0.0f;

  const int c = t;                       // this thread owns output column c
  for (int k = 0; k < F; ++k) {
    float lk = L[(size_t)k * F + c];     // coalesced across threads
#pragma unroll
    for (int r = 0; r < 16; ++r) acc[r] = fmaf(Xs[r][k], lk, acc[r]);  // broadcast LDS read
  }

  const int lane = t & 31;
#pragma unroll
  for (int r = 0; r < 16; ++r) {
    float v = acc[r];
    unsigned short hb = f32_to_bf16_bits(v);
    float hf = bf16_bits_to_f32(hb);
    unsigned short lb = f32_to_bf16_bits(v - hf);
    size_t o = (size_t)(rowBase + r) * F + c;
    dH[o] = hb; dL[o] = lb;
    // row norm contribution, reduced across the wave32 then one atomic
    float w = v * v;
    w += __shfl_xor(w, 1,  32);
    w += __shfl_xor(w, 2,  32);
    w += __shfl_xor(w, 4,  32);
    w += __shfl_xor(w, 8,  32);
    w += __shfl_xor(w, 16, 32);
    if (lane == 0) atomicAdd(&nrm[rowBase + r], w);
  }
}

// ---------------------------------------------------------------------------
// Kernel 2: Gram GEMM on v_wmma_f32_16x16x32_bf16 with split-bf16 emulation
// (hi*hi + hi*lo + lo*hi ~ f32 accuracy), double-buffered LDS fed by
// GLOBAL_LOAD_ASYNC_TO_LDS_B128 (ASYNCcnt-pipelined), fused sqrt/exp
// epilogue, per-row partial-sum atomics.
// Block = 256 threads (8 waves), tile BM x BN = 128 x 64, wave tile 32 x 32.
// ---------------------------------------------------------------------------
__global__ __launch_bounds__(256) void gram_kernel(
    const unsigned short* __restrict__ XLh, const unsigned short* __restrict__ XLl,
    const unsigned short* __restrict__ YLh, const unsigned short* __restrict__ YLl,
    const float* __restrict__ xnorm, const float* __restrict__ ynorm,
    float* __restrict__ out, float* __restrict__ rowsum,
    int n1, int n2)
{
  // 2 pipeline buffers x (Yh, Yl, Xh, Xl): 2 * 384 rows * 80B = 60 KB
  __shared__ __align__(16) unsigned short Ysh_h[2][BM][LROW];
  __shared__ __align__(16) unsigned short Ysh_l[2][BM][LROW];
  __shared__ __align__(16) unsigned short Xsh_h[2][BN][LROW];
  __shared__ __align__(16) unsigned short Xsh_l[2][BN][LROW];

  const int t    = threadIdx.x;
  const int lane = t & 31;
  const int wave = t >> 5;        // 0..7
  const int wy   = wave >> 1;     // 0..3 : i direction
  const int wx   = wave & 1;      // 0..1 : j direction
  const int rsel = lane & 15;     // M / N within 16x16 tile
  const int hsel = lane >> 4;     // half-wave select
  const int iBase = blockIdx.y * BM;
  const int jBase = blockIdx.x * BN;

  // Issue one K-step's async copies into buffer `buf`. Exactly 6 async ops
  // per wave (in-order completion), so s_wait_asynccnt 6 retires the
  // previously staged buffer while these remain in flight.
  auto stage = [&](int buf, int kk) {
    for (int n = t; n < BM * 4; n += 256) {       // 2 iterations
      int row = n >> 2, seg = n & 3;
      size_t go = (size_t)(iBase + row) * F + kk + seg * 8;
      unsigned lofs = (unsigned)((row * LROW + seg * 8) * 2);
      async_copy_b128((unsigned)(uintptr_t)&Ysh_h[buf][0][0] + lofs, YLh + go);
      async_copy_b128((unsigned)(uintptr_t)&Ysh_l[buf][0][0] + lofs, YLl + go);
    }
    for (int n = t; n < BN * 4; n += 256) {       // 1 iteration
      int row = n >> 2, seg = n & 3;
      size_t go = (size_t)(jBase + row) * F + kk + seg * 8;
      unsigned lofs = (unsigned)((row * LROW + seg * 8) * 2);
      async_copy_b128((unsigned)(uintptr_t)&Xsh_h[buf][0][0] + lofs, XLh + go);
      async_copy_b128((unsigned)(uintptr_t)&Xsh_l[buf][0][0] + lofs, XLl + go);
    }
  };

  v8f acc[2][2] = {};

  stage(0, 0);                    // prologue: fill buffer 0
  int cur = 0;

  for (int kk = 0; kk < F; kk += KT) {
    const bool hasNext = (kk + KT) < F;
    if (hasNext) {
      stage(cur ^ 1, kk + KT);    // prefetch next K-step into other buffer
      asm volatile("s_wait_asynccnt 0x6" ::: "memory");  // current buffer done
    } else {
      asm volatile("s_wait_asynccnt 0x0" ::: "memory");  // drain
    }
    __syncthreads();              // all waves' async writes to `cur` visible

    // A frag (16x32 bf16): lanes 0-15 -> M=0..15, K={0..7,16..23};
    //                      lanes 16-31 -> same M, K={8..15,24..31}
    Frag ah[2], al[2], bh[2], bl[2];
#pragma unroll
    for (int ti = 0; ti < 2; ++ti) {
      int row = wy * 32 + ti * 16 + rsel;
      int ka  = hsel * 8;
      ah[ti].q[0] = *(const uint4*)&Ysh_h[cur][row][ka];
      ah[ti].q[1] = *(const uint4*)&Ysh_h[cur][row][ka + 16];
      al[ti].q[0] = *(const uint4*)&Ysh_l[cur][row][ka];
      al[ti].q[1] = *(const uint4*)&Ysh_l[cur][row][ka + 16];
    }
    // B frag (32x16 bf16): lane -> N = lane&15; K = {8h..8h+7, 16+8h..16+8h+7}, h = lane>>4
#pragma unroll
    for (int tj = 0; tj < 2; ++tj) {
      int col = wx * 32 + tj * 16 + rsel;
      int kb  = hsel * 8;
      bh[tj].q[0] = *(const uint4*)&Xsh_h[cur][col][kb];
      bh[tj].q[1] = *(const uint4*)&Xsh_h[cur][col][kb + 16];
      bl[tj].q[0] = *(const uint4*)&Xsh_l[cur][col][kb];
      bl[tj].q[1] = *(const uint4*)&Xsh_l[cur][col][kb + 16];
    }

#pragma unroll
    for (int ti = 0; ti < 2; ++ti)
#pragma unroll
      for (int tj = 0; tj < 2; ++tj) {
        acc[ti][tj] = __builtin_amdgcn_wmma_f32_16x16x32_bf16(
            false, ah[ti].v, false, bh[tj].v, (short)0, acc[ti][tj], false, false);
        acc[ti][tj] = __builtin_amdgcn_wmma_f32_16x16x32_bf16(
            false, ah[ti].v, false, bl[tj].v, (short)0, acc[ti][tj], false, false);
        acc[ti][tj] = __builtin_amdgcn_wmma_f32_16x16x32_bf16(
            false, al[ti].v, false, bh[tj].v, (short)0, acc[ti][tj], false, false);
      }
    __syncthreads();              // everyone done reading `cur` before overwrite
    cur ^= 1;
  }

  // ---- fused epilogue: k = exp(-d/F)/sqrt(2*pi); write + row partial sums ----
  // C/D layout: vgpr r -> M = r + 8*hsel ; N = lane&15
  const size_t pitch = (size_t)n1;
#pragma unroll
  for (int ti = 0; ti < 2; ++ti) {
    float rs[8];
#pragma unroll
    for (int r = 0; r < 8; ++r) rs[r] = 0.0f;
    const int irow0 = iBase + wy * 32 + ti * 16 + hsel * 8;
#pragma unroll
    for (int tj = 0; tj < 2; ++tj) {
      int j = jBase + wx * 32 + tj * 16 + rsel;
      float xn = xnorm[j];
#pragma unroll
      for (int r = 0; r < 8; ++r) {
        int i = irow0 + r;
        float sq = ynorm[i] + xn - 2.0f * acc[ti][tj][r];
        float d  = sqrtf(fmaxf(sq, 0.0f));
        float kv = expf(d * (-1.0f / (float)F)) * 0.3989422804014327f;
        out[(size_t)i * pitch + j] = kv;
        rs[r] += kv;
      }
    }
    // reduce across each 16-lane half (rows differ between halves)
#pragma unroll
    for (int r = 0; r < 8; ++r) {
      rs[r] += __shfl_xor(rs[r], 1, 32);
      rs[r] += __shfl_xor(rs[r], 2, 32);
      rs[r] += __shfl_xor(rs[r], 4, 32);
      rs[r] += __shfl_xor(rs[r], 8, 32);
    }
    if (rsel == 0) {
#pragma unroll
      for (int r = 0; r < 8; ++r) atomicAdd(&rowsum[irow0 + r], rs[r]);
    }
  }
}

// ---------------------------------------------------------------------------
// Kernel 3: out[i,j] /= (rowsum[i] + 1e-6), float4 vectorized single pass.
// ---------------------------------------------------------------------------
__global__ __launch_bounds__(256) void normalize_kernel(
    float* __restrict__ out, const float* __restrict__ rowsum,
    int n1, size_t total)
{
  size_t idx = ((size_t)blockIdx.x * blockDim.x + threadIdx.x) * 4;
  if (idx >= total) return;
  float4 v = *(const float4*)(out + idx);
  size_t i = idx / (size_t)n1;          // n1 % 4 == 0 -> whole float4 in one row
  float inv = 1.0f / (rowsum[i] + 1e-6f);
  v.x *= inv; v.y *= inv; v.z *= inv; v.w *= inv;
  *(float4*)(out + idx) = v;
}

// ---------------------------------------------------------------------------
extern "C" void kernel_launch(void* const* d_in, const int* in_sizes, int n_in,
                              void* d_out, int out_size, void* d_ws, size_t ws_size,
                              hipStream_t stream)
{
  const float* X = (const float*)d_in[0];   // support_x [n1, F]
  const float* Y = (const float*)d_in[1];   // support_y [n2, F]
  const float* L = (const float*)d_in[2];   // L [F, F]
  const int n1 = in_sizes[0] / F;
  const int n2 = in_sizes[1] / F;
  float* out = (float*)d_out;               // [n2, n1]

  // workspace layout (all offsets multiples of 512B): ~16.1 MB total
  char* ws = (char*)d_ws;
  size_t o = 0;
  unsigned short* XLh = (unsigned short*)(ws + o); o += (size_t)n1 * F * 2;
  unsigned short* XLl = (unsigned short*)(ws + o); o += (size_t)n1 * F * 2;
  unsigned short* YLh = (unsigned short*)(ws + o); o += (size_t)n2 * F * 2;
  unsigned short* YLl = (unsigned short*)(ws + o); o += (size_t)n2 * F * 2;
  float* xnorm  = (float*)(ws + o); o += (size_t)n1 * 4;
  float* ynorm  = (float*)(ws + o); o += (size_t)n2 * 4;
  float* rowsum = (float*)(ws + o); o += (size_t)n2 * 4;

  // zero the atomic accumulators (norms + rowsums)
  hipMemsetAsync(xnorm, 0, (size_t)(n1 + 2 * n2) * 4, stream);

  const int maxr = n1 > n2 ? n1 : n2;
  dim3 pgrid((unsigned)((maxr + 15) / 16), 2);
  prep_gemm_split<<<pgrid, 256, 0, stream>>>(X, Y, L, XLh, XLl, YLh, YLl,
                                             xnorm, ynorm, n1, n2);

  dim3 ggrid((unsigned)(n1 / BN), (unsigned)(n2 / BM));
  gram_kernel<<<ggrid, 256, 0, stream>>>(XLh, XLl, YLh, YLl,
                                         xnorm, ynorm, out, rowsum, n1, n2);

  size_t total = (size_t)n1 * (size_t)n2;
  unsigned nblocks = (unsigned)((total / 4 + 255) / 256);
  normalize_kernel<<<nblocks, 256, 0, stream>>>(out, rowsum, n1, total);
}
